// CRFLayer_63127429316859
// MI455X (gfx1250) — matrix operand order repaired and checked by reading, not compiled, not run on hardware
//
#include <hip/hip_runtime.h>

typedef __attribute__((ext_vector_type(16))) _Float16 v16h;
typedef __attribute__((ext_vector_type(2)))  __fp16   v2fp;
typedef __attribute__((ext_vector_type(8)))  float    v8f;
typedef __attribute__((ext_vector_type(4)))  float    f4v;

#define Bn 512
#define Tn 2048
#define Kn 32

#define LOG2E 1.4426950408889634f
#define LN2   0.6931471805599453f

// Cross-half (lane ^ 16) exchange via v_permlanex16_b32 (VALU, no LDS).
__device__ __forceinline__ unsigned xh_u32(unsigned x) {
  return __builtin_amdgcn_permlanex16(x, x, 0x76543210u, 0xFEDCBA98u,
                                      false, false);
}
__device__ __forceinline__ float xh_f32(float x) {
  return __uint_as_float(xh_u32(__float_as_uint(x)));
}
// Pack two f32 -> f16x2 dword (v_cvt_pk_rtz_f16_f32).
__device__ __forceinline__ unsigned pk2(float a, float b) {
  union { v2fp h; unsigned u; } z;
  z.h = __builtin_amdgcn_cvt_pkrtz(a, b);
  return z.u;
}

union H8 { unsigned u[8]; v16h h; };

// ---------------------------------------------------------------------------
// Forward algorithm (denominator). One wave32 per 16 batches.
// alpha kept in log2 units:  a2'[j] = m2 + log2( sum_i 2^(a2[i]-m2) * E[i][j] )
//                                   + em[t][j]*log2(e)
// Inner sum: 16x16x32 f16 WMMA, D(tag,batch) = E^T(tag,i) x P(i,batch).
// ---------------------------------------------------------------------------
__global__ __launch_bounds__(32) void crf_forward_kernel(
    const float* __restrict__ emissions,    // [B,T,K]
    const int*   __restrict__ seqlen,       // [B]
    const float* __restrict__ startT,       // [K]
    const float* __restrict__ endT,         // [K]
    const float* __restrict__ trans,        // [K,K]
    float*       __restrict__ denom_out)    // [B]
{
  const int lane = threadIdx.x & 31;
  const int hi   = (lane >= 16) ? 1 : 0;
  const int bloc = lane & 15;
  const int batch = blockIdx.x * 16 + bloc;

  // A matrices (resident in VGPRs): A = E^T, E = exp(trans) (linear domain).
  // A k order used here (contiguous, k=16*half+elem, same as B): row M = lane&15; elems 0..7 -> K=klo..klo+7,
  // elems 8..15 -> K=khi..khi+7 (klo=0/16, khi=8/24 by lane half).
  const int klo = hi ? 16 : 0;
  const int khi = hi ? 24 : 8;
  v16h a0, a1;   // a0: tags j=0..15 ; a1: tags j=16..31
  {
    const int j0 = bloc, j1 = 16 + bloc;
#pragma unroll
    for (int e = 0; e < 8; ++e) {
      const int i = klo + e;
      a0[e] = (_Float16)__builtin_amdgcn_exp2f(trans[i * Kn + j0] * LOG2E);
      a1[e] = (_Float16)__builtin_amdgcn_exp2f(trans[i * Kn + j1] * LOG2E);
    }
#pragma unroll
    for (int e = 0; e < 8; ++e) {
      const int i = khi + e;
      a0[8 + e] = (_Float16)__builtin_amdgcn_exp2f(trans[i * Kn + j0] * LOG2E);
      a1[8 + e] = (_Float16)__builtin_amdgcn_exp2f(trans[i * Kn + j1] * LOG2E);
    }
  }

  // D/C layout: element r, lane<16 -> (tag=r, batch=lane); lane>=16 -> tag=8+r.
  const int off0 = hi ? 8 : 0;   // tag offset of d0-tile elements
  const int off1 = off0 + 16;    // tag offset of d1-tile elements

  const float* row = emissions + (size_t)batch * Tn * Kn;
  const int L = seqlen[batch];

  // alpha (log2 units) at t=0 : (start_transitions + emissions[:,0]) * log2e
  float aLo[8], aHi[8];
#pragma unroll
  for (int r = 0; r < 8; ++r) {
    aLo[r] = (startT[off0 + r] + row[off0 + r]) * LOG2E;
    aHi[r] = (startT[off1 + r] + row[off1 + r]) * LOG2E;
  }

  const v8f cz = {};

  for (int t = 1; t < Tn; ++t) {
    const float* erow = row + (size_t)t * Kn;
    f4v e0a = *(const f4v*)(erow + off0);
    f4v e0b = *(const f4v*)(erow + off0 + 4);
    f4v e1a = *(const f4v*)(erow + off1);
    f4v e1b = *(const f4v*)(erow + off1 + 4);
    __builtin_prefetch(erow + Kn, 0, 1);   // next step's row

    // m = max over this batch's 32 tags: tree + one cross-half swap
    float tm[8];
#pragma unroll
    for (int r = 0; r < 8; ++r) tm[r] = fmaxf(aLo[r], aHi[r]);
#pragma unroll
    for (int s = 4; s > 0; s >>= 1)
#pragma unroll
      for (int r = 0; r < 4; ++r)
        if (r < s) tm[r] = fmaxf(tm[r], tm[r + s]);
    const float m = fmaxf(tm[0], xh_f32(tm[0]));

    // p = 2^(alpha - m), packed into f16x2 dwords (bare v_exp_f32)
    float pLo[8], pHi[8];
#pragma unroll
    for (int r = 0; r < 8; ++r) {
      pLo[r] = __builtin_amdgcn_exp2f(aLo[r] - m);
      pHi[r] = __builtin_amdgcn_exp2f(aHi[r] - m);
    }
    unsigned pl[4], ph[4];
#pragma unroll
    for (int r = 0; r < 4; ++r) {
      pl[r] = pk2(pLo[2 * r], pLo[2 * r + 1]);
      ph[r] = pk2(pHi[2 * r], pHi[2 * r + 1]);
    }

    // B matrix (32x16 f16): lane<16 holds K=0..15 of batch=lane (own pLo +
    // partner pLo); lane>=16 holds K=16..31 (partner pHi + own pHi).
    // Each lane sends (hi ? pLo : pHi); one permlanex16 per dword.
    H8 bmu;
#pragma unroll
    for (int r = 0; r < 4; ++r) {
      const unsigned snd = hi ? pl[r] : ph[r];
      const unsigned rcv = xh_u32(snd);
      bmu.u[r]     = hi ? rcv   : pl[r];
      bmu.u[4 + r] = hi ? ph[r] : rcv;
    }

    // D(tag,batch) = E^T x P : two 16x16 tiles cover all 32 tags
    v8f d0 = __builtin_amdgcn_wmma_f32_16x16x32_f16(
        false, a0, false, bmu.h, (short)0, cz, false, false);
    v8f d1 = __builtin_amdgcn_wmma_f32_16x16x32_f16(
        false, a1, false, bmu.h, (short)0, cz, false, false);

    // me = m + em*log2e : independent of D, schedules under the WMMA
    float me0[8], me1[8];
#pragma unroll
    for (int r = 0; r < 4; ++r) {
      me0[r]     = fmaf(e0a[r], LOG2E, m);
      me0[4 + r] = fmaf(e0b[r], LOG2E, m);
      me1[r]     = fmaf(e1a[r], LOG2E, m);
      me1[4 + r] = fmaf(e1b[r], LOG2E, m);
    }

    // alpha' = me + log2(D) (bare v_log_f32) ; masked by t < L
    const bool keep = (t < L);
#pragma unroll
    for (int r = 0; r < 8; ++r) {
      const float n0 = me0[r] + __builtin_amdgcn_logf(d0[r]);
      const float n1 = me1[r] + __builtin_amdgcn_logf(d1[r]);
      if (keep) { aLo[r] = n0; aHi[r] = n1; }
    }
  }

  // denom_b = logsumexp(alpha_T + end_transitions), computed in log2 domain
  float x[8], y[8], tm[8];
#pragma unroll
  for (int r = 0; r < 8; ++r) {
    x[r] = fmaf(endT[off0 + r], LOG2E, aLo[r]);
    y[r] = fmaf(endT[off1 + r], LOG2E, aHi[r]);
    tm[r] = fmaxf(x[r], y[r]);
  }
#pragma unroll
  for (int s = 4; s > 0; s >>= 1)
#pragma unroll
    for (int r = 0; r < 4; ++r)
      if (r < s) tm[r] = fmaxf(tm[r], tm[r + s]);
  const float m2 = fmaxf(tm[0], xh_f32(tm[0]));
  float s = 0.f;
#pragma unroll
  for (int r = 0; r < 8; ++r)
    s += __builtin_amdgcn_exp2f(x[r] - m2) + __builtin_amdgcn_exp2f(y[r] - m2);
  s += xh_f32(s);
  if (!hi) denom_out[batch] = (m2 + __builtin_amdgcn_logf(s)) * LN2;
}

// ---------------------------------------------------------------------------
// Numerator score: one block per batch, deterministic tree reduction.
// ---------------------------------------------------------------------------
__global__ __launch_bounds__(256) void crf_score_kernel(
    const float* __restrict__ emissions,
    const int*   __restrict__ tags,
    const int*   __restrict__ seqlen,
    const float* __restrict__ startT,
    const float* __restrict__ endT,
    const float* __restrict__ trans,
    float*       __restrict__ score_out)
{
  __shared__ float red[256];
  const int b = blockIdx.x, tid = threadIdx.x;
  const int L = seqlen[b];
  const int*   tg  = tags + (size_t)b * Tn;
  const float* row = emissions + (size_t)b * Tn * Kn;

  float local = 0.f;
  for (int t = 1 + tid; t < L; t += 256) {
    const int tp = tg[t - 1], tc = tg[t];
    local += row[(size_t)t * Kn + tc] + trans[tp * Kn + tc];
  }
  if (tid == 0) {
    const int t0 = tg[0];
    local += startT[t0] + row[t0] + endT[tg[L - 1]];
  }
  red[tid] = local;
  __syncthreads();
#pragma unroll
  for (int s = 128; s > 0; s >>= 1) {
    if (tid < s) red[tid] += red[tid + s];
    __syncthreads();
  }
  if (tid == 0) score_out[b] = red[0];
}

// ---------------------------------------------------------------------------
// Final deterministic reduction: out = sum_b (score_b - denom_b)
// ---------------------------------------------------------------------------
__global__ __launch_bounds__(512) void crf_reduce_kernel(
    const float* __restrict__ score,
    const float* __restrict__ denom,
    float*       __restrict__ out)
{
  __shared__ float red[512];
  const int tid = threadIdx.x;
  red[tid] = score[tid] - denom[tid];
  __syncthreads();
#pragma unroll
  for (int s = 256; s > 0; s >>= 1) {
    if (tid < s) red[tid] += red[tid + s];
    __syncthreads();
  }
  if (tid == 0) out[0] = red[0];
}

extern "C" void kernel_launch(void* const* d_in, const int* in_sizes, int n_in,
                              void* d_out, int out_size, void* d_ws, size_t ws_size,
                              hipStream_t stream) {
  const float* emissions = (const float*)d_in[0];
  const int*   tags      = (const int*)  d_in[1];
  const int*   seqlen    = (const int*)  d_in[2];
  const float* startT    = (const float*)d_in[3];
  const float* endT      = (const float*)d_in[4];
  const float* trans     = (const float*)d_in[5];
  float* out = (float*)d_out;

  float* score = (float*)d_ws;        // [B]
  float* denom = score + Bn;          // [B]

  crf_score_kernel<<<Bn, 256, 0, stream>>>(emissions, tags, seqlen,
                                           startT, endT, trans, score);
  crf_forward_kernel<<<Bn / 16, 32, 0, stream>>>(emissions, seqlen,
                                                 startT, endT, trans, denom);
  crf_reduce_kernel<<<1, 512, 0, stream>>>(score, denom, out);
}
